// ATTPredictor_55035710931312
// MI455X (gfx1250) — matrix operand order repaired and checked by reading, not compiled, not run on hardware
//
#include <hip/hip_runtime.h>
#include <hip/hip_bf16.h>

// ---------------------------------------------------------------------------
// ATTPredictor for MI455X (gfx1250, wave32, WMMA + async-to-LDS staging).
// All matmul-shaped work goes through v_wmma_f32_16x16x32_bf16:
//   conv1/conv2 (implicit GEMM), LSTM x/h projections, emb_lin,
//   combine, GRU gi/gh, output projection.
// GEMM operand staging uses GLOBAL_LOAD_ASYNC_TO_LDS_B128 (ASYNCcnt) with
// double-buffered LDS so loads overlap WMMA.
// ---------------------------------------------------------------------------

typedef __attribute__((ext_vector_type(16))) __bf16 v16bf;
typedef __attribute__((ext_vector_type(8)))  float  v8f;

#define NB    256   // batch / rois
#define CCH   256   // channels
#define VOC   97
#define NUMCH 25
#define TT    64    // sequence length (W after convs)

__device__ __forceinline__ float sigm(float x) { return 1.0f / (1.0f + __expf(-x)); }

struct Frag { union { v16bf v; uint4 q[2]; }; };

// A fragment (16x32 bf16, M across lanes 0..15 / 16..31, interleaved K halves)
__device__ __forceinline__ v16bf load_frag_a(const __bf16* lds, int rowbase, int r16, int half) {
    Frag f;
    const uint4* p = (const uint4*)(lds + (rowbase + r16) * 32);
    f.q[0] = p[half];        // K = 8*half .. 8*half+7
    f.q[1] = p[2 + half];    // K = 16+8*half .. 16+8*half+7
    return f.v;
}
// B fragment (32x16 bf16 as Bt rows; lane n = r16, interleaved K halves)
__device__ __forceinline__ v16bf load_frag_b(const __bf16* lds, int rowbase, int r16, int half) {
    Frag f;
    const uint4* p = (const uint4*)(lds + (rowbase + r16) * 32);
    f.q[0] = p[half];
    f.q[1] = p[2 + half];
    return f.v;
}

__device__ __forceinline__ v8f wmma_bf16(v16bf a, v16bf b, v8f c) {
    return __builtin_amdgcn_wmma_f32_16x16x32_bf16(false, a, false, b, (short)0, c, false, false);
}

// LDS byte offset of a generic pointer known to point into LDS.
__device__ __forceinline__ unsigned lds_off(const void* p) {
    return (unsigned)(unsigned long long)
        (__attribute__((address_space(3))) const void*)p;
}

// Async copy 32B (2 x B128 lanes' worth is done by two calls of 16B each)
__device__ __forceinline__ void async_b128(unsigned lds, const void* gaddr) {
    asm volatile("global_load_async_to_lds_b128 %0, %1, off"
                 :: "v"(lds), "v"(gaddr) : "memory");
}
__device__ __forceinline__ void async_wait0() {
    asm volatile("s_wait_asynccnt 0x0" ::: "memory");
}

// ---------------------------------------------------------------------------
// Generic NT GEMM: C[M,N] = A[M,K] * B[N,K]^T (+bias, +relu)
// Block tile 128x128, K-step 32, 256 threads = 8 waves, wave tile 32x64.
// Interior tiles stream both operands via async-to-LDS, double buffered.
// ---------------------------------------------------------------------------
__global__ __launch_bounds__(256) void k_gemm_nt(
    const __bf16* __restrict__ A, const __bf16* __restrict__ B,
    const float* __restrict__ bias, float* __restrict__ Cf, __bf16* __restrict__ Cb,
    int M, int N, int K, int lda, int ldb, int ldcf, int ldcb, int relu)
{
    __shared__ __bf16 sA[2][128 * 32];
    __shared__ __bf16 sB[2][128 * 32];
    const int tid  = threadIdx.x;
    const int wave = tid >> 5, lane = tid & 31;
    const int half = lane >> 4, r16 = lane & 15;
    const int bm = blockIdx.x * 128, bn = blockIdx.y * 128;
    const int wm = (wave & 3) * 32, wn = (wave >> 2) * 64;

    const v8f vzero = {0.f,0.f,0.f,0.f,0.f,0.f,0.f,0.f};
    v8f acc[2][4];
    #pragma unroll
    for (int i = 0; i < 2; i++)
        #pragma unroll
        for (int j = 0; j < 4; j++) acc[i][j] = vzero;

    const int row = tid >> 1, col = (tid & 1) * 16;   // 128 rows x 32B each
    const bool interior = (bm + 128 <= M) && (bn + 128 <= N);

    auto stage = [&](int k0, int buf) {
        if (interior) {
            const __bf16* ga = A + (size_t)(bm + row) * lda + k0 + col;
            const __bf16* gb = B + (size_t)(bn + row) * ldb + k0 + col;
            async_b128(lds_off(&sA[buf][row * 32 + col]),     ga);
            async_b128(lds_off(&sA[buf][row * 32 + col + 8]), (const __bf16*)ga + 8);
            async_b128(lds_off(&sB[buf][row * 32 + col]),     gb);
            async_b128(lds_off(&sB[buf][row * 32 + col + 8]), (const __bf16*)gb + 8);
        } else {
            uint4 a0 = make_uint4(0u,0u,0u,0u), a1 = a0;
            int m = bm + row;
            if (m < M) {
                const uint4* pa = (const uint4*)(A + (size_t)m * lda + k0 + col);
                a0 = pa[0]; a1 = pa[1];
            }
            *(uint4*)(&sA[buf][row * 32 + col])     = a0;
            *(uint4*)(&sA[buf][row * 32 + col + 8]) = a1;
            uint4 b0 = make_uint4(0u,0u,0u,0u), b1 = b0;
            int n = bn + row;
            if (n < N) {
                const uint4* pb = (const uint4*)(B + (size_t)n * ldb + k0 + col);
                b0 = pb[0]; b1 = pb[1];
            }
            *(uint4*)(&sB[buf][row * 32 + col])     = b0;
            *(uint4*)(&sB[buf][row * 32 + col + 8]) = b1;
        }
    };

    stage(0, 0);
    async_wait0();
    __syncthreads();

    int cur = 0;
    for (int k0 = 0; k0 < K; k0 += 32) {
        int nk = k0 + 32;
        if (nk < K) stage(nk, cur ^ 1);

        v16bf fa0 = load_frag_a(&sA[cur][0], wm,      r16, half);
        v16bf fa1 = load_frag_a(&sA[cur][0], wm + 16, r16, half);
        v16bf fb0 = load_frag_b(&sB[cur][0], wn,      r16, half);
        v16bf fb1 = load_frag_b(&sB[cur][0], wn + 16, r16, half);
        v16bf fb2 = load_frag_b(&sB[cur][0], wn + 32, r16, half);
        v16bf fb3 = load_frag_b(&sB[cur][0], wn + 48, r16, half);
        acc[0][0] = wmma_bf16(fa0, fb0, acc[0][0]);
        acc[0][1] = wmma_bf16(fa0, fb1, acc[0][1]);
        acc[0][2] = wmma_bf16(fa0, fb2, acc[0][2]);
        acc[0][3] = wmma_bf16(fa0, fb3, acc[0][3]);
        acc[1][0] = wmma_bf16(fa1, fb0, acc[1][0]);
        acc[1][1] = wmma_bf16(fa1, fb1, acc[1][1]);
        acc[1][2] = wmma_bf16(fa1, fb2, acc[1][2]);
        acc[1][3] = wmma_bf16(fa1, fb3, acc[1][3]);

        async_wait0();
        __syncthreads();
        cur ^= 1;
    }

    for (int mt = 0; mt < 2; mt++)
    for (int nt = 0; nt < 4; nt++) {
        int ncol = bn + wn + nt * 16 + r16;
        if (ncol >= N) continue;
        float bv = bias ? bias[ncol] : 0.0f;
        #pragma unroll
        for (int rr = 0; rr < 8; rr++) {
            int mrow = bm + wm + mt * 16 + 8 * half + rr;
            if (mrow >= M) continue;
            float v = acc[mt][nt][rr] + bv;
            if (relu) v = fmaxf(v, 0.0f);
            if (Cf) Cf[(size_t)mrow * ldcf + ncol] = v;
            if (Cb) Cb[(size_t)mrow * ldcb + ncol] = (__bf16)v;
        }
    }
}

// ---------------------------------------------------------------------------
// Implicit-GEMM conv: 3x3, stride (2,1), pad (1,1), Cin=Cout=256, NHWC bf16.
// M = NB*Hout*Wout, N = 256, K = 9*256 = 2304; weights repacked (o,(ky,kx,c)).
// Weights stream via async-to-LDS; im2col A-gather uses guarded VGPR path.
// ---------------------------------------------------------------------------
__global__ __launch_bounds__(256) void k_conv_gemm(
    const __bf16* __restrict__ in, const __bf16* __restrict__ Wp, __bf16* __restrict__ out,
    int Hin, int Win, int Hout, int Wout, int M)
{
    __shared__ __bf16 sA[2][128 * 32];
    __shared__ __bf16 sB[2][128 * 32];
    const int tid  = threadIdx.x;
    const int wave = tid >> 5, lane = tid & 31;
    const int half = lane >> 4, r16 = lane & 15;
    const int bm = blockIdx.x * 128, bn = blockIdx.y * 128;
    const int wm = (wave & 3) * 32, wn = (wave >> 2) * 64;
    const int K = 9 * CCH;

    const v8f vzero = {0.f,0.f,0.f,0.f,0.f,0.f,0.f,0.f};
    v8f acc[2][4];
    #pragma unroll
    for (int i = 0; i < 2; i++)
        #pragma unroll
        for (int j = 0; j < 4; j++) acc[i][j] = vzero;

    const int row = tid >> 1, col = (tid & 1) * 16;
    // Decode this thread's output pixel once.
    const int m = bm + row;
    int wo = m % Wout, tmp = m / Wout;
    int ho = tmp % Hout, nimg = tmp / Hout;

    auto stage = [&](int k0, int buf) {
        // weights: always interior -> async
        const __bf16* gb = Wp + (size_t)(bn + row) * K + k0 + col;
        async_b128(lds_off(&sB[buf][row * 32 + col]),     gb);
        async_b128(lds_off(&sB[buf][row * 32 + col + 8]), gb + 8);
        // im2col gather with zero padding
        int kk  = k0 + col;
        int kc  = kk & 255;
        int kxy = kk >> 8;
        int ky = kxy / 3, kx = kxy - 3 * ky;
        uint4 a0 = make_uint4(0u,0u,0u,0u), a1 = a0;
        if (m < M) {
            int h = 2 * ho - 1 + ky;
            int w = wo - 1 + kx;
            if (h >= 0 && h < Hin && w >= 0 && w < Win) {
                const uint4* pa = (const uint4*)
                    (in + ((((size_t)nimg * Hin + h) * Win + w) << 8) + kc);
                a0 = pa[0]; a1 = pa[1];
            }
        }
        *(uint4*)(&sA[buf][row * 32 + col])     = a0;
        *(uint4*)(&sA[buf][row * 32 + col + 8]) = a1;
    };

    stage(0, 0);
    async_wait0();
    __syncthreads();

    int cur = 0;
    for (int k0 = 0; k0 < K; k0 += 32) {
        int nk = k0 + 32;
        if (nk < K) stage(nk, cur ^ 1);

        v16bf fa0 = load_frag_a(&sA[cur][0], wm,      r16, half);
        v16bf fa1 = load_frag_a(&sA[cur][0], wm + 16, r16, half);
        v16bf fb0 = load_frag_b(&sB[cur][0], wn,      r16, half);
        v16bf fb1 = load_frag_b(&sB[cur][0], wn + 16, r16, half);
        v16bf fb2 = load_frag_b(&sB[cur][0], wn + 32, r16, half);
        v16bf fb3 = load_frag_b(&sB[cur][0], wn + 48, r16, half);
        acc[0][0] = wmma_bf16(fa0, fb0, acc[0][0]);
        acc[0][1] = wmma_bf16(fa0, fb1, acc[0][1]);
        acc[0][2] = wmma_bf16(fa0, fb2, acc[0][2]);
        acc[0][3] = wmma_bf16(fa0, fb3, acc[0][3]);
        acc[1][0] = wmma_bf16(fa1, fb0, acc[1][0]);
        acc[1][1] = wmma_bf16(fa1, fb1, acc[1][1]);
        acc[1][2] = wmma_bf16(fa1, fb2, acc[1][2]);
        acc[1][3] = wmma_bf16(fa1, fb3, acc[1][3]);

        async_wait0();
        __syncthreads();
        cur ^= 1;
    }

    for (int mt = 0; mt < 2; mt++)
    for (int nt = 0; nt < 4; nt++) {
        int ncol = bn + wn + nt * 16 + r16;
        #pragma unroll
        for (int rr = 0; rr < 8; rr++) {
            int mrow = bm + wm + mt * 16 + 8 * half + rr;
            if (mrow >= M) continue;
            out[(size_t)mrow * CCH + ncol] = (__bf16)acc[mt][nt][rr];
        }
    }
}

// ---------------------------------------------------------------------------
// Layout / conversion helpers
// ---------------------------------------------------------------------------
__global__ __launch_bounds__(256) void k_f32_to_bf16(const float* __restrict__ in,
                                                     __bf16* __restrict__ out, int n)
{
    int i = blockIdx.x * 256 + threadIdx.x;
    if (i < n) out[i] = (__bf16)in[i];
}

__global__ __launch_bounds__(256) void k_nchw_to_nhwc(const float* __restrict__ in,
                                                      __bf16* __restrict__ out, int H, int W)
{
    int b = blockIdx.x, c = threadIdx.x;
    int w = b % W; int t = b / W;
    int h = t % H; int n = t / H;
    out[(size_t)b * CCH + c] = (__bf16)in[(((size_t)n * CCH + c) * H + h) * W + w];
}

// OIHW (256,256,3,3) -> (o, (ky,kx,c)) bf16
__global__ __launch_bounds__(256) void k_conv_w_repack(const float* __restrict__ w,
                                                       __bf16* __restrict__ out)
{
    int o = blockIdx.x, ci = threadIdx.x;
    for (int ky = 0; ky < 3; ky++)
        for (int kx = 0; kx < 3; kx++)
            out[(size_t)o * 2304 + (ky * 3 + kx) * CCH + ci] =
                (__bf16)w[(((size_t)o * CCH + ci) * 3 + ky) * 3 + kx];
}

// ---------------------------------------------------------------------------
// GroupNorm (G=32, 8 ch/group) over NHWC bf16 activations (in-place) + ReLU
// ---------------------------------------------------------------------------
__global__ __launch_bounds__(256) void k_gn_stat(const __bf16* __restrict__ act,
                                                 float* __restrict__ stat, int HW)
{
    int n = blockIdx.x >> 5, g = blockIdx.x & 31;
    int tid = threadIdx.x;
    float s = 0.f, s2 = 0.f;
    for (int r = tid; r < HW; r += 256) {
        const __bf16* p = act + ((size_t)(n * HW + r)) * CCH + g * 8;
        #pragma unroll
        for (int j = 0; j < 8; j++) { float x = (float)p[j]; s += x; s2 += x * x; }
    }
    __shared__ float sa[256], sb[256];
    sa[tid] = s; sb[tid] = s2; __syncthreads();
    for (int st = 128; st > 0; st >>= 1) {
        if (tid < st) { sa[tid] += sa[tid + st]; sb[tid] += sb[tid + st]; }
        __syncthreads();
    }
    if (tid == 0) {
        float cnt = (float)(HW * 8);
        float mu = sa[0] / cnt;
        float var = sb[0] / cnt - mu * mu;
        stat[blockIdx.x * 2]     = mu;
        stat[blockIdx.x * 2 + 1] = rsqrtf(var + 1e-5f);
    }
}

__global__ __launch_bounds__(256) void k_gn_apply(__bf16* __restrict__ act,
    const float* __restrict__ stat, const float* __restrict__ sc,
    const float* __restrict__ bi, int HW)
{
    int m = blockIdx.x, c = threadIdx.x;
    int n = m / HW, g = c >> 3;
    float mu = stat[(n * 32 + g) * 2], rs = stat[(n * 32 + g) * 2 + 1];
    float x = (float)act[(size_t)m * CCH + c];
    x = (x - mu) * rs * sc[c] + bi[c];
    act[(size_t)m * CCH + c] = (__bf16)fmaxf(x, 0.0f);
}

// mean over H (=4) of act2 (N,4,64,256) -> seq (T=64, N, C) bf16
__global__ __launch_bounds__(256) void k_mean_h(const __bf16* __restrict__ act2,
                                                __bf16* __restrict__ seq)
{
    int b = blockIdx.x;          // (t<<8)|n
    int n = b & 255, t = b >> 8;
    int c = threadIdx.x;
    float s = 0.f;
    #pragma unroll
    for (int h = 0; h < 4; h++)
        s += (float)act2[(((size_t)(n * 4 + h) * 64 + t)) * CCH + c];
    seq[((size_t)(t * NB + n)) * CCH + c] = (__bf16)(s * 0.25f);
}

// ---------------------------------------------------------------------------
// LSTM cell: gates = xproj(t) + hproj + bih + bhh ; split i,f,g,o
// ---------------------------------------------------------------------------
__global__ __launch_bounds__(256) void k_lstm_cell(
    const float* __restrict__ xp, const float* __restrict__ hp,
    const float* __restrict__ bih, const float* __restrict__ bhh,
    float* __restrict__ cst, __bf16* __restrict__ hbf, __bf16* __restrict__ rec_t)
{
    int n = blockIdx.x, c = threadIdx.x;
    size_t b4 = (size_t)n * 1024;
    float gi = xp[b4 + c]       + hp[b4 + c]       + bih[c]       + bhh[c];
    float gf = xp[b4 + 256 + c] + hp[b4 + 256 + c] + bih[256 + c] + bhh[256 + c];
    float gg = xp[b4 + 512 + c] + hp[b4 + 512 + c] + bih[512 + c] + bhh[512 + c];
    float go = xp[b4 + 768 + c] + hp[b4 + 768 + c] + bih[768 + c] + bhh[768 + c];
    float cc = sigm(gf) * cst[n * CCH + c] + sigm(gi) * tanhf(gg);
    float hh = sigm(go) * tanhf(cc);
    cst[n * CCH + c] = cc;
    hbf[n * CCH + c] = (__bf16)hh;
    rec_t[(size_t)n * 512 + c] = (__bf16)hh;
}

// ---------------------------------------------------------------------------
// Attention + context + decoder-input build (one block per roi n)
// ---------------------------------------------------------------------------
__global__ __launch_bounds__(256) void k_attn(
    const float* __restrict__ enc, const float* __restrict__ hid,
    const float* __restrict__ vat_w, const float* __restrict__ vat_b,
    const __bf16* __restrict__ emb_tab, const int* __restrict__ dec_in,
    __bf16* __restrict__ acat)
{
    __shared__ __bf16 s_enc[TT * CCH];
    __shared__ float  s_h[CCH], s_vw[CCH], s_sc[TT];
    int n = blockIdx.x, c = threadIdx.x;
    s_h[c]  = hid[n * CCH + c];
    s_vw[c] = vat_w[c];
    for (int t = 0; t < TT; t++)
        s_enc[t * CCH + c] = (__bf16)enc[((size_t)(t * NB + n)) * CCH + c];
    __syncthreads();

    int wave = c >> 5, lane = c & 31;
    for (int t = wave; t < TT; t += 8) {
        float s = 0.f;
        for (int cc = lane; cc < CCH; cc += 32)
            s += tanhf(s_h[cc] + (float)s_enc[t * CCH + cc]) * s_vw[cc];
        #pragma unroll
        for (int m = 16; m > 0; m >>= 1) s += __shfl_xor(s, m, 32);
        if (lane == 0) s_sc[t] = s + vat_b[0];
    }
    __syncthreads();
    if (c == 0) {
        float mx = -1e30f;
        for (int t = 0; t < TT; t++) mx = fmaxf(mx, s_sc[t]);
        float sm = 0.f;
        for (int t = 0; t < TT; t++) { float e = __expf(s_sc[t] - mx); s_sc[t] = e; sm += e; }
        float inv = 1.0f / sm;
        for (int t = 0; t < TT; t++) s_sc[t] *= inv;
    }
    __syncthreads();
    float ctx = 0.f;
    for (int t = 0; t < TT; t++) ctx += s_sc[t] * (float)s_enc[t * CCH + c];
    acat[(size_t)n * 512 + 256 + c] = (__bf16)ctx;
    acat[(size_t)n * 512 + c] = emb_tab[dec_in[n] * CCH + c];
}

// ---------------------------------------------------------------------------
// GRU cell: r,z,n gates -> new hidden (f32 + bf16 copies)
// ---------------------------------------------------------------------------
__global__ __launch_bounds__(256) void k_gru_cell(
    const float* __restrict__ gi, const float* __restrict__ gh,
    float* __restrict__ hid, __bf16* __restrict__ hid_bf)
{
    int n = blockIdx.x, c = threadIdx.x;
    size_t b3 = (size_t)n * 768;
    float r  = sigm(gi[b3 + c]       + gh[b3 + c]);
    float z  = sigm(gi[b3 + 256 + c] + gh[b3 + 256 + c]);
    float nn = tanhf(gi[b3 + 512 + c] + r * gh[b3 + 512 + c]);
    float h  = hid[n * CCH + c];
    h = (1.0f - z) * nn + z * h;
    hid[n * CCH + c] = h;
    hid_bf[n * CCH + c] = (__bf16)h;
}

// ---------------------------------------------------------------------------
// log-softmax over 97 logits, NLL accumulation, greedy argmax feedback
// ---------------------------------------------------------------------------
__global__ __launch_bounds__(128) void k_loss_argmax(
    const float* __restrict__ logits, const int* __restrict__ targets, int step,
    float* __restrict__ loss, int* __restrict__ dec_in)
{
    int n = blockIdx.x, t = threadIdx.x;
    __shared__ float sv[128]; __shared__ int si[128]; __shared__ float se[128];
    float v = (t < VOC) ? logits[n * VOC + t] : -1e30f;
    sv[t] = v; si[t] = t; __syncthreads();
    for (int s = 64; s > 0; s >>= 1) {
        if (t < s) {
            if (sv[t + s] > sv[t] || (sv[t + s] == sv[t] && si[t + s] < si[t])) {
                sv[t] = sv[t + s]; si[t] = si[t + s];
            }
        }
        __syncthreads();
    }
    float mx = sv[0]; int amax = si[0]; __syncthreads();
    se[t] = (t < VOC) ? __expf(logits[n * VOC + t] - mx) : 0.f;
    __syncthreads();
    for (int s = 64; s > 0; s >>= 1) {
        if (t < s) se[t] += se[t + s];
        __syncthreads();
    }
    if (t == 0) {
        float logZ = mx + logf(se[0]);
        int tgt = targets[n * NUMCH + step];
        atomicAdd(loss, -(logits[n * VOC + tgt] - logZ) * (1.0f / (float)NB));
        dec_in[n] = amax;
    }
}

// ---------------------------------------------------------------------------
// Host launcher
// ---------------------------------------------------------------------------
extern "C" void kernel_launch(void* const* d_in, const int* in_sizes, int n_in,
                              void* d_out, int out_size, void* d_ws, size_t ws_size,
                              hipStream_t stream)
{
    (void)in_sizes; (void)n_in; (void)out_size; (void)ws_size;
    const float* rois     = (const float*)d_in[0];
    const int*   targets  = (const int*)  d_in[1];
    const float* conv1_w  = (const float*)d_in[2];
    const float* gn1_s    = (const float*)d_in[3];
    const float* gn1_b    = (const float*)d_in[4];
    const float* conv2_w  = (const float*)d_in[5];
    const float* gn2_s    = (const float*)d_in[6];
    const float* gn2_b    = (const float*)d_in[7];
    const float* wih_f    = (const float*)d_in[8];
    const float* whh_f    = (const float*)d_in[9];
    const float* bih_f    = (const float*)d_in[10];
    const float* bhh_f    = (const float*)d_in[11];
    const float* wih_b    = (const float*)d_in[12];
    const float* whh_b    = (const float*)d_in[13];
    const float* bih_b    = (const float*)d_in[14];
    const float* bhh_b    = (const float*)d_in[15];
    const float* embw     = (const float*)d_in[16];
    const float* embb     = (const float*)d_in[17];
    const float* embtab   = (const float*)d_in[18];
    const float* combw    = (const float*)d_in[19];
    const float* combb    = (const float*)d_in[20];
    const float* gruwih   = (const float*)d_in[21];
    const float* gruwhh   = (const float*)d_in[22];
    const float* grubih   = (const float*)d_in[23];
    const float* grubhh   = (const float*)d_in[24];
    const float* outw     = (const float*)d_in[25];
    const float* outb     = (const float*)d_in[26];
    const float* vatw     = (const float*)d_in[27];
    const float* vatb     = (const float*)d_in[28];
    float* loss = (float*)d_out;

    // ---- workspace carve ----
    char* ws = (char*)d_ws;
    size_t off = 0;
    auto alloc = [&](size_t bytes) -> char* {
        char* p = ws + off;
        off = (off + bytes + 255) & ~(size_t)255;
        return p;
    };
    // big region: rois_nhwc (134 MB) later reused by xproj_f/xproj_b (67 MB each)
    char*   big       = alloc((size_t)134217728);
    __bf16* rois_nhwc = (__bf16*)big;
    float*  xproj_f   = (float*)big;
    float*  xproj_b   = (float*)(big + 67108864);

    __bf16* w1p   = (__bf16*)alloc((size_t)256 * 2304 * 2);
    __bf16* w2p   = (__bf16*)alloc((size_t)256 * 2304 * 2);
    __bf16* act1  = (__bf16*)alloc((size_t)131072 * 256 * 2);
    __bf16* act2  = (__bf16*)alloc((size_t)65536 * 256 * 2);
    __bf16* seq   = (__bf16*)alloc((size_t)TT * NB * CCH * 2);
    __bf16* rec   = (__bf16*)alloc((size_t)TT * NB * 512 * 2);
    float*  enc   = (float*)alloc((size_t)TT * NB * CCH * 4);

    __bf16* wihf_bf  = (__bf16*)alloc(262144 * 2);
    __bf16* whhf_bf  = (__bf16*)alloc(262144 * 2);
    __bf16* wihb_bf  = (__bf16*)alloc(262144 * 2);
    __bf16* whhb_bf  = (__bf16*)alloc(262144 * 2);
    __bf16* embw_bf  = (__bf16*)alloc(131072 * 2);
    __bf16* embtab_bf= (__bf16*)alloc(24832 * 2);
    __bf16* combw_bf = (__bf16*)alloc(131072 * 2);
    __bf16* gwih_bf  = (__bf16*)alloc(196608 * 2);
    __bf16* gwhh_bf  = (__bf16*)alloc(196608 * 2);
    __bf16* outw_bf  = (__bf16*)alloc(24832 * 2);

    float*  gnstat = (float*)alloc(8192 * 2 * 4);
    float*  hproj  = (float*)alloc((size_t)256 * 1024 * 4);
    __bf16* h_f    = (__bf16*)alloc(65536 * 2);
    __bf16* h_b    = (__bf16*)alloc(65536 * 2);
    float*  c_f    = (float*)alloc(65536 * 4);
    float*  c_b    = (float*)alloc(65536 * 4);
    float*  hid    = (float*)alloc(65536 * 4);
    __bf16* hid_bf = (__bf16*)alloc(65536 * 2);
    __bf16* acat   = (__bf16*)alloc((size_t)256 * 512 * 2);
    __bf16* comb_bf= (__bf16*)alloc(65536 * 2);
    float*  gi     = (float*)alloc((size_t)256 * 768 * 4);
    float*  gh     = (float*)alloc((size_t)256 * 768 * 4);
    float*  logits = (float*)alloc((size_t)256 * VOC * 4);
    int*    dec_in = (int*)alloc(256 * 4);

    // ---- init ----
    hipMemsetAsync(loss, 0, 4, stream);

    // ---- weight conversions ----
    auto cvt = [&](const float* src, __bf16* dst, int n) {
        k_f32_to_bf16<<<(n + 255) / 256, 256, 0, stream>>>(src, dst, n);
    };
    cvt(wih_f, wihf_bf, 262144);  cvt(whh_f, whhf_bf, 262144);
    cvt(wih_b, wihb_bf, 262144);  cvt(whh_b, whhb_bf, 262144);
    cvt(embw, embw_bf, 131072);   cvt(embtab, embtab_bf, 24832);
    cvt(combw, combw_bf, 131072);
    cvt(gruwih, gwih_bf, 196608); cvt(gruwhh, gwhh_bf, 196608);
    cvt(outw, outw_bf, 24832);
    k_conv_w_repack<<<256, 256, 0, stream>>>(conv1_w, w1p);
    k_conv_w_repack<<<256, 256, 0, stream>>>(conv2_w, w2p);

    // ---- CRNN ----
    k_nchw_to_nhwc<<<NB * 16 * 64, 256, 0, stream>>>(rois, rois_nhwc, 16, 64);
    k_conv_gemm<<<dim3(131072 / 128, 2), 256, 0, stream>>>(rois_nhwc, w1p, act1, 16, 64, 8, 64, 131072);
    k_gn_stat<<<8192, 256, 0, stream>>>(act1, gnstat, 512);
    k_gn_apply<<<131072, 256, 0, stream>>>(act1, gnstat, gn1_s, gn1_b, 512);
    k_conv_gemm<<<dim3(65536 / 128, 2), 256, 0, stream>>>(act1, w2p, act2, 8, 64, 4, 64, 65536);
    k_gn_stat<<<8192, 256, 0, stream>>>(act2, gnstat, 256);
    k_gn_apply<<<65536, 256, 0, stream>>>(act2, gnstat, gn2_s, gn2_b, 256);
    k_mean_h<<<TT * NB, 256, 0, stream>>>(act2, seq);

    // ---- bidirectional LSTM ----
    k_gemm_nt<<<dim3(128, 8), 256, 0, stream>>>(seq, wihf_bf, nullptr, xproj_f, nullptr,
                                                TT * NB, 1024, 256, 256, 256, 1024, 0, 0);
    k_gemm_nt<<<dim3(128, 8), 256, 0, stream>>>(seq, wihb_bf, nullptr, xproj_b, nullptr,
                                                TT * NB, 1024, 256, 256, 256, 1024, 0, 0);
    hipMemsetAsync(h_f, 0, 65536 * 2, stream);
    hipMemsetAsync(h_b, 0, 65536 * 2, stream);
    hipMemsetAsync(c_f, 0, 65536 * 4, stream);
    hipMemsetAsync(c_b, 0, 65536 * 4, stream);
    for (int t = 0; t < TT; t++) {
        k_gemm_nt<<<dim3(2, 8), 256, 0, stream>>>(h_f, whhf_bf, nullptr, hproj, nullptr,
                                                  256, 1024, 256, 256, 256, 1024, 0, 0);
        k_lstm_cell<<<256, 256, 0, stream>>>(xproj_f + (size_t)t * NB * 1024, hproj,
                                             bih_f, bhh_f, c_f, h_f,
                                             rec + (size_t)t * NB * 512);
    }
    for (int s = 0; s < TT; s++) {
        int t = TT - 1 - s;
        k_gemm_nt<<<dim3(2, 8), 256, 0, stream>>>(h_b, whhb_bf, nullptr, hproj, nullptr,
                                                  256, 1024, 256, 256, 256, 1024, 0, 0);
        k_lstm_cell<<<256, 256, 0, stream>>>(xproj_b + (size_t)t * NB * 1024, hproj,
                                             bih_b, bhh_b, c_b, h_b,
                                             rec + (size_t)t * NB * 512 + 256);
    }

    // ---- encoder linear: enc = rec @ emb_lin_w^T + b ----
    k_gemm_nt<<<dim3(128, 2), 256, 0, stream>>>(rec, embw_bf, embb, enc, nullptr,
                                                TT * NB, 256, 512, 512, 512, 256, 0, 0);

    // ---- attention decoder ----
    hipMemsetAsync(hid, 0, 65536 * 4, stream);
    hipMemsetAsync(hid_bf, 0, 65536 * 2, stream);
    hipMemsetAsync(dec_in, 0, 256 * 4, stream);
    for (int step = 0; step < NUMCH; step++) {
        k_attn<<<256, 256, 0, stream>>>(enc, hid, vatw, vatb, embtab_bf, dec_in, acat);
        k_gemm_nt<<<dim3(2, 2), 256, 0, stream>>>(acat, combw_bf, combb, nullptr, comb_bf,
                                                  256, 256, 512, 512, 512, 0, 256, 1);
        k_gemm_nt<<<dim3(2, 6), 256, 0, stream>>>(comb_bf, gwih_bf, grubih, gi, nullptr,
                                                  256, 768, 256, 256, 256, 768, 0, 0);
        k_gemm_nt<<<dim3(2, 6), 256, 0, stream>>>(hid_bf, gwhh_bf, grubhh, gh, nullptr,
                                                  256, 768, 256, 256, 256, 768, 0, 0);
        k_gru_cell<<<256, 256, 0, stream>>>(gi, gh, hid, hid_bf);
        k_gemm_nt<<<dim3(2, 1), 256, 0, stream>>>(hid_bf, outw_bf, outb, logits, nullptr,
                                                  256, VOC, 256, 256, 256, VOC, 0, 0);
        k_loss_argmax<<<256, 128, 0, stream>>>(logits, targets, step, loss, dec_in);
    }
}
